// MoE_layer_49735721288142
// MI455X (gfx1250) — matrix operand order repaired and checked by reading, not compiled, not run on hardware
//
#include <hip/hip_runtime.h>
#include <hip/hip_bf16.h>

typedef __attribute__((ext_vector_type(16))) __bf16          v16bf;
typedef __attribute__((ext_vector_type(16))) unsigned short  v16us;
typedef __attribute__((ext_vector_type(8)))  unsigned short  v8us;
typedef __attribute__((ext_vector_type(8)))  float           v8f;

#define NTOK 16384
#define DDIM 1024
#define HDIM 4096
#define NEXP 8
#define PROJDIM 256
#define MTILE 32
#define CLAMP_MAX 4.605170185988091f   // log(1/0.01)

__device__ __forceinline__ unsigned short f32_to_bf16(float f) {
  unsigned int u = __float_as_uint(f);
  unsigned int r = (u + 0x7FFFu + ((u >> 16) & 1u)) >> 16;   // RNE
  return (unsigned short)r;
}
__device__ __forceinline__ unsigned int pack2_bf16(float a, float b) {
  return (unsigned int)f32_to_bf16(a) | ((unsigned int)f32_to_bf16(b) << 16);
}

// ---------------------------------------------------------------- gate ------
__global__ void zero_counts_kernel(int* counts) {
  if (threadIdx.x < NEXP) counts[threadIdx.x] = 0;
}

__global__ void __launch_bounds__(256) gate_kernel(
    const float* __restrict__ x, const float* __restrict__ Wp,
    const float* __restrict__ bp, const float* __restrict__ sim,
    const float* __restrict__ temp, int* __restrict__ counts,
    int* __restrict__ tok_list, float* __restrict__ gate_list) {
  __shared__ float sx[DDIM];
  __shared__ float sp[PROJDIM];
  __shared__ float red[256];
  __shared__ float slog[NEXP];
  const int n = blockIdx.x;
  const int t = threadIdx.x;

  for (int i = t; i < DDIM; i += 256) sx[i] = x[(size_t)n * DDIM + i];
  __syncthreads();

  float acc = 0.f;
  for (int d = 0; d < DDIM; ++d) acc = fmaf(sx[d], Wp[d * PROJDIM + t], acc);
  acc += bp[t];
  sp[t]  = acc;
  red[t] = acc * acc;
  __syncthreads();
  for (int s = 128; s > 0; s >>= 1) {
    if (t < s) red[t] += red[t + s];
    __syncthreads();
  }
  float nrm = fmaxf(sqrtf(red[0]), 1e-12f);
  sp[t] = sp[t] / nrm;
  __syncthreads();

  if (t < NEXP) {
    float dot = 0.f, ss = 0.f;
    for (int p = 0; p < PROJDIM; ++p) {
      float sv = sim[p * NEXP + t];
      dot = fmaf(sp[p], sv, dot);
      ss  = fmaf(sv, sv, ss);
    }
    float cn    = fmaxf(sqrtf(ss), 1e-12f);
    float scale = expf(fminf(temp[0], CLAMP_MAX));
    slog[t] = (dot / cn) * scale;
  }
  __syncthreads();

  if (t == 0) {
    float v0 = -1e30f, v1 = -1e30f; int i0 = 0, i1 = 0;
    for (int e = 0; e < NEXP; ++e) {           // ties -> lowest index (jax top_k)
      float v = slog[e];
      if (v > v0)      { v1 = v0; i1 = i0; v0 = v; i0 = e; }
      else if (v > v1) { v1 = v;  i1 = e; }
    }
    float ex = expf(v1 - v0);
    float g0 = 1.f / (1.f + ex);
    float g1 = ex  / (1.f + ex);
    int p0 = atomicAdd(&counts[i0], 1);
    tok_list[i0 * NTOK + p0]  = n;
    gate_list[i0 * NTOK + p0] = g0;
    int p1 = atomicAdd(&counts[i1], 1);
    tok_list[i1 * NTOK + p1]  = n;
    gate_list[i1 * NTOK + p1] = g1;
  }
}

// ---------------------------------------------------- residual init --------
__global__ void copy_kernel(const float4* __restrict__ src,
                            float4* __restrict__ dst, int n4) {
  int i = blockIdx.x * blockDim.x + threadIdx.x;
  if (i < n4) dst[i] = src[i];
}

// ------------------------------------------------------------ expert FFN ----
// 256 threads = 8 waves; one block per (expert, 32-assignment tile).
// Weight tiles live in LDS K-major-per-column: w_t[c*32 + k], so a lane's
// 16x16x32 B fragment is two aligned ds_load_b128s. X/H tiles are row-major,
// A fragments are likewise two aligned ds_load_b128s.
__global__ void __launch_bounds__(256) ffn_kernel(
    const float* __restrict__ x,  const float* __restrict__ W1,
    const float* __restrict__ b1, const float* __restrict__ W2,
    const float* __restrict__ b2, const int* __restrict__ counts,
    const int* __restrict__ tok_list, const float* __restrict__ gate_list,
    float* __restrict__ out) {
  extern __shared__ char smem[];
  unsigned short* x_t = (unsigned short*)smem;          // 32x1024 row-major (64KB)
  unsigned short* w_t = x_t + MTILE * DDIM;             // 1024 cols x 32 k  (64KB)
  unsigned short* h_t = w_t + 32 * DDIM;                // 32x128 row-major  (8KB)
  float* g_s   = (float*)(h_t + MTILE * 128);           // 32 gates
  int*   tok_s = (int*)(g_s + MTILE);                   // 32 tokens

  const int tiles = NTOK / MTILE;                       // 512
  const int e    = blockIdx.x / tiles;
  const int tile = blockIdx.x % tiles;
  const int cnt  = counts[e];
  const int base = tile * MTILE;
  if (base >= cnt) return;                              // uniform exit

  const int tid  = threadIdx.x;
  const int wave = tid >> 5;
  const int lane = tid & 31;
  const int ln16 = lane & 15;
  const int hi   = lane >> 4;
  const int koff  = hi ? 8 : 0;                         // A-frag K offset
  const int kbase = hi ? 8 : 0;                         // B-frag K offset

  if (tid < MTILE) {
    int a = base + tid;
    if (a < cnt) { tok_s[tid] = tok_list[e * NTOK + a]; g_s[tid] = gate_list[e * NTOK + a]; }
    else         { tok_s[tid] = 0;                      g_s[tid] = 0.f; }
  }
  __syncthreads();

  // stage X tile: float4 global loads -> packed bf16 pairs -> ds_store_b64
  for (int i = tid; i < MTILE * DDIM / 4; i += 256) {
    int r  = i >> 8;                                    // 256 float4 per row
    int c4 = i & 255;
    float4 v = ((const float4*)(x + (size_t)tok_s[r] * DDIM))[c4];
    uint2 p;
    p.x = pack2_bf16(v.x, v.y);
    p.y = pack2_bf16(v.z, v.w);
    ((uint2*)x_t)[i] = p;
  }

  const float* W1e = W1 + (size_t)e * DDIM * HDIM;
  const float* W2e = W2 + (size_t)e * HDIM * DDIM;
  const float* b1e = b1 + (size_t)e * HDIM;
  const float* b2e = b2 + (size_t)e * DDIM;

  const v8f z8 = {0.f, 0.f, 0.f, 0.f, 0.f, 0.f, 0.f, 0.f};
  v8f facc[2][8];
#pragma unroll
  for (int rb = 0; rb < 2; ++rb)
#pragma unroll
    for (int j = 0; j < 8; ++j) facc[rb][j] = z8;

  for (int hc = 0; hc < HDIM; hc += 128) {
    v8f hacc[2] = {z8, z8};

    // ---- FFN1: H-tile(32x128) += X(32x1024) @ W1[:, hc:hc+128] ----
    for (int kd = 0; kd < DDIM; kd += 32) {
      __syncthreads();
      // stage W1 block 32k x 128c, K-major per column, one b128 store per unit
      for (int u = tid; u < 128 * 4; u += 256) {
        int c  = u & 127;
        int kg = u >> 7;                                // kgroup of 8
        const float* srcp = W1e + (size_t)(kd + kg * 8) * HDIM + hc + c;
        v8us pk;
#pragma unroll
        for (int j = 0; j < 8; ++j) pk[j] = f32_to_bf16(srcp[(size_t)j * HDIM]);
        *(v8us*)&w_t[c * 32 + kg * 8] = pk;
      }
      __syncthreads();

      const unsigned short* bp0 = &w_t[(wave * 16 + ln16) * 32 + kbase];
      v8us blo = *(const v8us*)(bp0);
      v8us bhi = *(const v8us*)(bp0 + 16);
      v16bf bf = __builtin_bit_cast(v16bf, __builtin_shufflevector(
          blo, bhi, 0, 1, 2, 3, 4, 5, 6, 7, 8, 9, 10, 11, 12, 13, 14, 15));

#pragma unroll
      for (int rb = 0; rb < 2; ++rb) {
        const unsigned short* ap = &x_t[(rb * 16 + ln16) * DDIM + kd + koff];
        v8us alo = *(const v8us*)(ap);
        v8us ahi = *(const v8us*)(ap + 16);
        v16bf af = __builtin_bit_cast(v16bf, __builtin_shufflevector(
            alo, ahi, 0, 1, 2, 3, 4, 5, 6, 7, 8, 9, 10, 11, 12, 13, 14, 15));
        hacc[rb] = __builtin_amdgcn_wmma_f32_16x16x32_bf16(
            false, af, false, bf, (short)0, hacc[rb], false, false);
      }
    }
    __syncthreads();

    // ---- bias + exact gelu -> h_t (bf16, row-major) ----
#pragma unroll
    for (int rb = 0; rb < 2; ++rb)
#pragma unroll
      for (int r = 0; r < 8; ++r) {
        int rowM = rb * 16 + r + (hi ? 8 : 0);
        int hcol = wave * 16 + ln16;
        float h  = hacc[rb][r] + b1e[hc + hcol];
        float gl = 0.5f * h * (1.f + erff(h * 0.70710678118654752f));
        h_t[rowM * 128 + hcol] = f32_to_bf16(gl);
      }

    // ---- FFN2: F(32x1024) += gelu(H)(32x128) @ W2[hc:hc+128, :] ----
    for (int kk = 0; kk < 128; kk += 32) {
      __syncthreads();
      // stage W2 block 32k x 1024c, K-major per column
      for (int u = tid; u < 1024 * 4; u += 256) {
        int c  = u & 1023;
        int kg = u >> 10;
        const float* srcp = W2e + (size_t)(hc + kk + kg * 8) * DDIM + c;
        v8us pk;
#pragma unroll
        for (int j = 0; j < 8; ++j) pk[j] = f32_to_bf16(srcp[(size_t)j * DDIM]);
        *(v8us*)&w_t[c * 32 + kg * 8] = pk;
      }
      if (hc + kk + 32 < HDIM)
        __builtin_prefetch(W2e + (size_t)(hc + kk + 32) * DDIM, 0, 1);
      __syncthreads();

      v16bf af2[2];
#pragma unroll
      for (int rb = 0; rb < 2; ++rb) {
        const unsigned short* ap = &h_t[(rb * 16 + ln16) * 128 + kk + koff];
        v8us alo = *(const v8us*)(ap);
        v8us ahi = *(const v8us*)(ap + 16);
        af2[rb] = __builtin_bit_cast(v16bf, __builtin_shufflevector(
            alo, ahi, 0, 1, 2, 3, 4, 5, 6, 7, 8, 9, 10, 11, 12, 13, 14, 15));
      }
#pragma unroll
      for (int j = 0; j < 8; ++j) {
        const unsigned short* bp2 = &w_t[(wave * 128 + j * 16 + ln16) * 32 + kbase];
        v8us blo = *(const v8us*)(bp2);
        v8us bhi = *(const v8us*)(bp2 + 16);
        v16bf bf2 = __builtin_bit_cast(v16bf, __builtin_shufflevector(
            blo, bhi, 0, 1, 2, 3, 4, 5, 6, 7, 8, 9, 10, 11, 12, 13, 14, 15));
#pragma unroll
        for (int rb = 0; rb < 2; ++rb)
          facc[rb][j] = __builtin_amdgcn_wmma_f32_16x16x32_bf16(
              false, af2[rb], false, bf2, (short)0, facc[rb][j], false, false);
      }
    }
  }

  // ---- scatter: out[token] += g * (F + b2) ----
#pragma unroll
  for (int rb = 0; rb < 2; ++rb)
#pragma unroll
    for (int j = 0; j < 8; ++j)
#pragma unroll
      for (int r = 0; r < 8; ++r) {
        int rowM = rb * 16 + r + (hi ? 8 : 0);
        int col  = wave * 128 + j * 16 + ln16;
        float g  = g_s[rowM];
        if (g != 0.f) {
          float v = g * (facc[rb][j][r] + b2e[col]);
          atomicAdd(&out[(size_t)tok_s[rowM] * DDIM + col], v);
        }
      }
}

// ---------------------------------------------------------------- launch ----
extern "C" void kernel_launch(void* const* d_in, const int* in_sizes, int n_in,
                              void* d_out, int out_size, void* d_ws, size_t ws_size,
                              hipStream_t stream) {
  (void)in_sizes; (void)n_in; (void)out_size; (void)ws_size;
  const float* x    = (const float*)d_in[0];
  const float* Wp   = (const float*)d_in[1];
  const float* bp   = (const float*)d_in[2];
  const float* sim  = (const float*)d_in[3];
  const float* temp = (const float*)d_in[4];
  const float* W1   = (const float*)d_in[5];
  const float* b1   = (const float*)d_in[6];
  const float* W2   = (const float*)d_in[7];
  const float* b2   = (const float*)d_in[8];
  float* out = (float*)d_out;

  char* ws         = (char*)d_ws;
  int*   counts    = (int*)ws;                                   // 8 ints
  int*   tok_list  = (int*)(ws + 256);                           // E*N ints
  float* gate_list = (float*)(ws + 256 + (size_t)NEXP * NTOK * sizeof(int));

  zero_counts_kernel<<<1, 32, 0, stream>>>(counts);
  gate_kernel<<<NTOK, 256, 0, stream>>>(x, Wp, bp, sim, temp,
                                        counts, tok_list, gate_list);
  int n4 = NTOK * DDIM / 4;
  copy_kernel<<<(n4 + 255) / 256, 256, 0, stream>>>((const float4*)x,
                                                    (float4*)out, n4);
  size_t shmem = (size_t)(MTILE * DDIM + 32 * DDIM) * sizeof(unsigned short)
               + (size_t)MTILE * 128 * sizeof(unsigned short)
               + MTILE * sizeof(float) + MTILE * sizeof(int);
  ffn_kernel<<<NEXP * (NTOK / MTILE), 256, shmem, stream>>>(
      x, W1, b1, W2, b2, counts, tok_list, gate_list, out);
}
